// PathMLP_65919158059528
// MI455X (gfx1250) — matrix operand order repaired and checked by reading, not compiled, not run on hardware
//
#include <hip/hip_runtime.h>
#include <math.h>

// ---------------------------------------------------------------------------
// PathMLP fused pipeline for gfx1250 (MI455X), wave32 + WMMA bf16 + async LDS.
//   P0a: pack W_proj|W_x -> bf16 Wc[192][128]; W_path -> bf16 Wp[128][256]
//   P0b: WA_T[i][h] = BETA * W_A[h][i]   (LDS-tiled transpose)
//   K_A: WMMA bf16 GEMM: pp_bf16[N][64] = x@W_proj.T+b ; h_f32[N][128]=gelu(x@W_x.T+b)
//   K_B: per 16 nodes: async-gather A[128][256] from pp via index straight to LDS,
//        WMMA GEMM vs W_path, gelu, dot-attention softmax over P=8, aggregate,
//        tmp = (1-B)*(agg+h) + B*b_A
//   K_C: edge scatter: tmp[row] += BETA*W_A.T[col]  (f32 atomics, L2 resident)
//   K_D: out = gelu(tmp) @ W_end.T + b_end
// ---------------------------------------------------------------------------

typedef __bf16 bf16;
typedef __bf16 v16bf __attribute__((ext_vector_type(16)));
typedef __bf16 v8bf  __attribute__((ext_vector_type(8)));
typedef __bf16 v4bf  __attribute__((ext_vector_type(4)));
typedef float  v8f   __attribute__((ext_vector_type(8)));

#define F_DIM   128
#define PH_DIM  64
#define P_DIM   8
#define D_DIM   4
#define DPH     256   // D_DIM*PH_DIM
#define H_DIM   128
#define C_DIM   10
#define NC_OUT  192   // PH_DIM + H_DIM (fused pp|h output columns)
#define BETA_C  0.5f

__device__ __forceinline__ float gelu_exact(float x) {
    return 0.5f * x * (1.0f + erff(x * 0.70710678118654752f));
}

__device__ __forceinline__ v8f wmma_bf16(v16bf a, v16bf b, v8f c) {
    // v_wmma_f32_16x16x32_bf16  D = A(16x32) * B(32x16) + C(16x16 f32)
    return __builtin_amdgcn_wmma_f32_16x16x32_bf16(
        /*neg_a=*/false, a, /*neg_b=*/false, b,
        /*c_mod=*/(short)0, c, /*reuse_a=*/false, /*reuse_b=*/false);
}

// Async 16B/lane copy global -> LDS (GLOBAL_LOAD_ASYNC_TO_LDS_B128, ASYNCcnt).
// Generic LDS pointer: addr[31:0] is the LDS byte offset (ISA 10.2 aperture).
__device__ __forceinline__ void async_copy_b128(void* lds_dst, const void* gsrc) {
    unsigned lds_off = (unsigned)(uintptr_t)lds_dst;
    asm volatile("global_load_async_to_lds_b128 %0, %1, off"
                 :: "v"(lds_off), "v"(gsrc) : "memory");
}
__device__ __forceinline__ void async_wait0() {
    asm volatile("s_wait_asynccnt 0x0" ::: "memory");
}

// A fragment (16x32 bf16): lanes 0-15 -> M=lane, K 0..7 then 16..23;
// lanes 16-31 -> M=lane-16, K 8..15 then 24..31.   [CDNA5 ISA]
__device__ __forceinline__ v16bf load_a_frag(const bf16* base, int row,
                                             int ldk, int k0, int lane) {
    int m  = lane & 15;
    int hi = lane >> 4;           // 0 or 1
    const bf16* p = base + (row + m) * ldk + k0 + hi * 8;
    union { v16bf v; v8bf h[2]; } u;
    u.h[0] = *(const v8bf*)(p);        // K +0..7   (ds_load_b128)
    u.h[1] = *(const v8bf*)(p + 16);   // K +16..23 (ds_load_b128)
    return u.v;
}

// B fragment (32x16 bf16): lane = N (mod 16); lanes 0-15 hold K 0..7 then 16..23,
// lanes 16-31 hold K 8..15 then 24..31. LDS stores W row-major [out_n][k] so K contiguous.
__device__ __forceinline__ v16bf load_b_frag(const bf16* base, int colrow0,
                                             int ldk, int k0, int lane) {
    int n  = lane & 15;
    int hi = lane >> 4;
    const bf16* p = base + (colrow0 + n) * ldk + k0 + hi * 8;
    union { v16bf v; v8bf h[2]; } u;
    u.h[0] = *(const v8bf*)(p);
    u.h[1] = *(const v8bf*)(p + 16);
    return u.v;
}

// --------------------------- P0a: weight packing ---------------------------
__global__ void prep_weights(const float* __restrict__ Wproj,
                             const float* __restrict__ Wx,
                             const float* __restrict__ Wpath,
                             bf16* __restrict__ Wc,     // [192][128]
                             bf16* __restrict__ Wp) {   // [128][256]
    int g = blockIdx.x * blockDim.x + threadIdx.x;
    const int nWc = NC_OUT * F_DIM;           // 24576
    const int nWp = H_DIM * DPH;              // 32768
    if (g < nWc) {
        int j = g / F_DIM, k = g % F_DIM;
        float v = (j < PH_DIM) ? Wproj[j * F_DIM + k]
                               : Wx[(j - PH_DIM) * F_DIM + k];
        Wc[g] = (bf16)v;
    } else if (g < nWc + nWp) {
        int t = g - nWc;
        Wp[t] = (bf16)Wpath[t];
    }
}

// --------------------------- P0b: beta * W_A.T -----------------------------
__global__ void transpose_wa(const float* __restrict__ WA, // [128][N]
                             float* __restrict__ WAT,      // [N][128]
                             int N) {
    __shared__ float tile[32][33];
    int tx = threadIdx.x, ty = threadIdx.y;      // 32 x 8
    int ib = blockIdx.x * 32, hb = blockIdx.y * 32;
#pragma unroll
    for (int r = 0; r < 4; ++r) {
        int h = hb + ty + r * 8;
        int i = ib + tx;
        tile[ty + r * 8][tx] = (i < N) ? WA[(size_t)h * N + i] : 0.0f;
    }
    __syncthreads();
#pragma unroll
    for (int r = 0; r < 4; ++r) {
        int i = ib + ty + r * 8;
        int h = hb + tx;
        if (i < N) WAT[(size_t)i * H_DIM + h] = BETA_C * tile[tx][ty + r * 8];
    }
}

// ------------------- K_A: fused pp/h GEMM (bf16 WMMA) ----------------------
// block = 256 thr (8 waves), 128 node-rows/block, 192 out cols, K=128.
// dyn LDS: XT 128*128 bf16 (32 KB) + WCs 192*128 bf16 (48 KB) = 80 KB.
__global__ void __launch_bounds__(256)
gemm_pp_h(const float* __restrict__ x, const bf16* __restrict__ Wc,
          const float* __restrict__ bproj, const float* __restrict__ bx,
          bf16* __restrict__ pp, float* __restrict__ hbuf, int N) {
    extern __shared__ char smem[];
    bf16* XT  = (bf16*)smem;                       // [128][128]
    bf16* WCs = (bf16*)(smem + 128 * 128 * 2);     // [192][128]

    const int tid  = threadIdx.x;
    const int lane = tid & 31;
    const int wv   = tid >> 5;
    const int row0 = blockIdx.x * 128;

    // async-stage packed weights (49152 B = 3072 x b128), no VGPR round-trip
#pragma unroll
    for (int it = 0; it < 12; ++it) {
        int e = tid + it * 256;
        async_copy_b128((uint4*)WCs + e, (const uint4*)Wc + e);
    }
    // load x tile (f32 -> bf16 conversion must pass through VGPRs)
#pragma unroll
    for (int it = 0; it < 16; ++it) {
        int g4  = tid + it * 256;          // 0..4095
        int r   = g4 >> 5;                 // local row
        int c4  = (g4 & 31) * 4;
        int gr  = row0 + r; if (gr >= N) gr = N - 1;
        float4 v = *(const float4*)(x + (size_t)gr * F_DIM + c4);
        v4bf b; b.x = (bf16)v.x; b.y = (bf16)v.y; b.z = (bf16)v.z; b.w = (bf16)v.w;
        *(v4bf*)(XT + r * F_DIM + c4) = b;
    }
    async_wait0();
    __syncthreads();

    const int rbase = wv * 16;
    v8f acc[12];
#pragma unroll
    for (int j = 0; j < 12; ++j) acc[j] = (v8f)0.0f;

#pragma unroll
    for (int ks = 0; ks < 4; ++ks) {
        int k0 = ks * 32;
        v16bf af = load_a_frag(XT, rbase, F_DIM, k0, lane);
        v16bf bfr[6];
#pragma unroll
        for (int half = 0; half < 2; ++half) {
#pragma unroll
            for (int j = 0; j < 6; ++j)
                bfr[j] = load_b_frag(WCs, (half * 6 + j) * 16, F_DIM, k0, lane);
#pragma unroll
            for (int j = 0; j < 6; ++j)
                acc[half * 6 + j] = wmma_bf16(af, bfr[j], acc[half * 6 + j]);
        }
    }

    // epilogue: C/D layout -> lane holds (m = v + 8*hi, n = lane&15)
    const int nsel = lane & 15;
    const int hi   = lane >> 4;
#pragma unroll
    for (int j = 0; j < 12; ++j) {
#pragma unroll
        for (int v = 0; v < 8; ++v) {
            int gr  = row0 + rbase + v + hi * 8;
            int col = j * 16 + nsel;
            if (gr < N) {
                float val = acc[j][v];
                if (col < PH_DIM) {
                    pp[(size_t)gr * PH_DIM + col] = (bf16)(val + bproj[col]);
                } else {
                    int hc = col - PH_DIM;
                    hbuf[(size_t)gr * H_DIM + hc] = gelu_exact(val + bx[hc]);
                }
            }
        }
    }
}

// ---------- K_B: fused path-gather GEMM + attention (bf16 WMMA) ------------
// block = 256 thr (8 waves) handles 16 nodes -> [128 x 256]@[256 x 128].
// dyn LDS: region0 64 KB (A tile bf16 128x256, later pi f32 128x128)
//          region1 64 KB (W_path bf16 128x256)     total 128 KB.
// Gather + weight staging use GLOBAL_LOAD_ASYNC_TO_LDS_B128 (ASYNCcnt).
__global__ void __launch_bounds__(256)
path_attn(const bf16* __restrict__ pp, const bf16* __restrict__ Wp,
          const int* __restrict__ index, const float* __restrict__ hbuf,
          const float* __restrict__ bpath, const float* __restrict__ bA,
          float* __restrict__ tmp, int N) {
    extern __shared__ char smem[];
    bf16*  Atile = (bf16*)smem;                      // [128][256] bf16
    float* piL   = (float*)smem;                     // overlay [128][128] f32
    bf16*  Wps   = (bf16*)(smem + 128 * 256 * 2);    // [128][256] bf16
    __shared__ float h_s[16 * H_DIM];
    __shared__ float att_s[16 * P_DIM];

    const int tid  = threadIdx.x;
    const int lane = tid & 31;
    const int wv   = tid >> 5;
    const int n0   = blockIdx.x * 16;

    // async gather A tile: row r=(nl*8+p), seg d: pp[index[n0+nl,p,d]][0..63]
    // 512 (r,d) pairs x 128 B, straight to LDS.
#pragma unroll
    for (int q = tid; q < 512; q += 256) {
        int r = q >> 2, d = q & 3;
        int nl = r >> 3, p = r & 7;
        int idx = index[(size_t)(n0 + nl) * (P_DIM * D_DIM) + p * D_DIM + d];
        const uint4* src = (const uint4*)(pp + (size_t)idx * PH_DIM); // 128 B
        uint4* dst = (uint4*)(Atile + r * DPH + d * PH_DIM);
#pragma unroll
        for (int u = 0; u < 8; ++u) async_copy_b128(dst + u, src + u);
    }
    // async-stage W_path (64 KB = 4096 x b128)
#pragma unroll
    for (int it = 0; it < 16; ++it) {
        int e = tid + it * 256;
        async_copy_b128((uint4*)Wps + e, (const uint4*)Wp + e);
    }
    // stage h rows for this node group (2048 f32, lives in VGPR path anyway)
#pragma unroll
    for (int it = 0; it < 8; ++it) {
        int e = tid + it * 256;
        h_s[e] = hbuf[(size_t)n0 * H_DIM + e];
    }
    async_wait0();
    __syncthreads();

    // GEMM: wave w -> rows [16w,16w+16), all 128 cols, K=256.
    // Batch all 8 B-fragments per k-step so ds_loads pipeline under WMMAs
    // (partial s_wait_dscnt instead of a full drain per WMMA).
    const int rbase = wv * 16;
    v8f acc[8];
#pragma unroll
    for (int j = 0; j < 8; ++j) acc[j] = (v8f)0.0f;
#pragma unroll
    for (int ks = 0; ks < 8; ++ks) {
        int k0 = ks * 32;
        v16bf af = load_a_frag(Atile, rbase, DPH, k0, lane);
        v16bf bfr[8];
#pragma unroll
        for (int j = 0; j < 8; ++j)
            bfr[j] = load_b_frag(Wps, j * 16, DPH, k0, lane);
#pragma unroll
        for (int j = 0; j < 8; ++j)
            acc[j] = wmma_bf16(af, bfr[j], acc[j]);
    }
    __syncthreads();   // all waves done reading Atile before overlay

    // epilogue: pi = gelu(acc + b_path) into overlay LDS (f32)
    const int nsel = lane & 15;
    const int hi   = lane >> 4;
#pragma unroll
    for (int j = 0; j < 8; ++j) {
#pragma unroll
        for (int v = 0; v < 8; ++v) {
            int row = rbase + v + hi * 8;
            int col = j * 16 + nsel;
            piL[row * H_DIM + col] = gelu_exact(acc[j][v] + bpath[col]);
        }
    }
    __syncthreads();

    // attention logits: thread (nl,p) -> dot(h[nl], pi[nl,p])
    if (tid < 128) {
        int nl = tid >> 3, p = tid & 7;
        const float* hv = h_s + nl * H_DIM;
        const float* pv = piL + (nl * P_DIM + p) * H_DIM;
        float s = 0.0f;
#pragma unroll 8
        for (int k = 0; k < H_DIM; ++k) s += hv[k] * pv[k];
        att_s[tid] = s;
    }
    __syncthreads();
    // softmax over P=8 per node
    if (tid < 16) {
        float* a = att_s + tid * P_DIM;
        float m = a[0];
#pragma unroll
        for (int p = 1; p < P_DIM; ++p) m = fmaxf(m, a[p]);
        float sum = 0.0f;
#pragma unroll
        for (int p = 0; p < P_DIM; ++p) { float e = __expf(a[p] - m); a[p] = e; sum += e; }
        float inv = 1.0f / sum;
#pragma unroll
        for (int p = 0; p < P_DIM; ++p) a[p] *= inv;
    }
    __syncthreads();
    // aggregate + write tmp = (1-B)*(agg+h) + B*b_A
    {
        int nl = tid >> 4;
        int c0 = (tid & 15) * 8;
        const float* a = att_s + nl * P_DIM;
#pragma unroll
        for (int c = c0; c < c0 + 8; ++c) {
            float s = 0.0f;
#pragma unroll
            for (int p = 0; p < P_DIM; ++p)
                s += a[p] * piL[(nl * P_DIM + p) * H_DIM + c];
            float o = (1.0f - BETA_C) * (s + h_s[nl * H_DIM + c]) + BETA_C * bA[c];
            tmp[(size_t)(n0 + nl) * H_DIM + c] = o;
        }
    }
}

// ------------------- K_C: edge scatter (f32 L2 atomics) --------------------
__global__ void __launch_bounds__(256)
edge_scatter(const int* __restrict__ ei, const float* __restrict__ WAT,
             float* __restrict__ tmp, int E) {
    int wave  = (blockIdx.x * blockDim.x + threadIdx.x) >> 5;
    int lane  = threadIdx.x & 31;
    int nwave = (gridDim.x * blockDim.x) >> 5;
    for (int e = wave; e < E; e += nwave) {
        int r = ei[e];          // destination segment
        int c = ei[E + e];      // gathered row
        float4 v = *(const float4*)(WAT + (size_t)c * H_DIM + lane * 4);
        float* dst = tmp + (size_t)r * H_DIM + lane * 4;
        atomicAdd(dst + 0, v.x);
        atomicAdd(dst + 1, v.y);
        atomicAdd(dst + 2, v.z);
        atomicAdd(dst + 3, v.w);
    }
}

// ---------------------- K_D: final gelu + W_end GEMV -----------------------
__global__ void __launch_bounds__(256)
final_out(const float* __restrict__ tmp, const float* __restrict__ Wend,
          const float* __restrict__ bend, float* __restrict__ out, int N) {
    __shared__ float We[C_DIM * H_DIM];
    __shared__ float Be[C_DIM];
    for (int i = threadIdx.x; i < C_DIM * H_DIM; i += 256) We[i] = Wend[i];
    if (threadIdx.x < C_DIM) Be[threadIdx.x] = bend[threadIdx.x];
    __syncthreads();
    int n = blockIdx.x * blockDim.x + threadIdx.x;
    if (n >= N) return;
    float acc[C_DIM];
#pragma unroll
    for (int c = 0; c < C_DIM; ++c) acc[c] = Be[c];
    const float* row = tmp + (size_t)n * H_DIM;
#pragma unroll 4
    for (int k = 0; k < H_DIM; ++k) {
        float v = gelu_exact(row[k]);
#pragma unroll
        for (int c = 0; c < C_DIM; ++c) acc[c] += v * We[c * H_DIM + k];
    }
#pragma unroll
    for (int c = 0; c < C_DIM; ++c) out[(size_t)n * C_DIM + c] = acc[c];
}

// ---------------------------------------------------------------------------
extern "C" void kernel_launch(void* const* d_in, const int* in_sizes, int n_in,
                              void* d_out, int out_size, void* d_ws, size_t ws_size,
                              hipStream_t stream) {
    const float* x     = (const float*)d_in[0];
    const float* Wproj = (const float*)d_in[1];
    const float* bproj = (const float*)d_in[2];
    const float* Wx    = (const float*)d_in[3];
    const float* bx    = (const float*)d_in[4];
    const float* Wpath = (const float*)d_in[5];
    const float* bpath = (const float*)d_in[6];
    const float* WA    = (const float*)d_in[7];
    const float* bA    = (const float*)d_in[8];
    const float* Wend  = (const float*)d_in[9];
    const float* bend  = (const float*)d_in[10];
    const int*   index = (const int*)d_in[11];
    const int*   ei    = (const int*)d_in[12];
    float* out = (float*)d_out;

    const int N = in_sizes[0] / F_DIM;       // 50000
    const int E = in_sizes[12] / 2;          // 1600000

    // workspace layout
    char* ws = (char*)d_ws;
    size_t off = 0;
    bf16*  pp   = (bf16*)(ws + off);  off += (size_t)N * PH_DIM * sizeof(bf16);   // 6.4 MB
    off = (off + 255) & ~(size_t)255;
    float* hbuf = (float*)(ws + off); off += (size_t)N * H_DIM * sizeof(float);   // 25.6 MB
    float* tmp  = (float*)(ws + off); off += (size_t)N * H_DIM * sizeof(float);   // 25.6 MB
    float* WAT  = (float*)(ws + off); off += (size_t)N * H_DIM * sizeof(float);   // 25.6 MB
    bf16*  Wc   = (bf16*)(ws + off);  off += (size_t)NC_OUT * F_DIM * sizeof(bf16);
    off = (off + 255) & ~(size_t)255;
    bf16*  Wp   = (bf16*)(ws + off);  off += (size_t)H_DIM * DPH * sizeof(bf16);

    const size_t ldsA = 128 * 128 * 2 + 192 * 128 * 2;   // 80 KB
    const size_t ldsB = 128 * 256 * 2 + 128 * 256 * 2;   // 128 KB
    hipFuncSetAttribute((const void*)gemm_pp_h,
                        hipFuncAttributeMaxDynamicSharedMemorySize, (int)ldsA);
    hipFuncSetAttribute((const void*)path_attn,
                        hipFuncAttributeMaxDynamicSharedMemorySize, (int)ldsB);

    // P0a: weight packing
    {
        int total = NC_OUT * F_DIM + H_DIM * DPH;
        prep_weights<<<(total + 255) / 256, 256, 0, stream>>>(Wproj, Wx, Wpath, Wc, Wp);
    }
    // P0b: beta * W_A.T
    {
        dim3 grid((N + 31) / 32, H_DIM / 32);
        dim3 block(32, 8);
        transpose_wa<<<grid, block, 0, stream>>>(WA, WAT, N);
    }
    // K_A: pp + h
    gemm_pp_h<<<(N + 127) / 128, 256, ldsA, stream>>>(x, Wc, bproj, bx, pp, hbuf, N);
    // K_B: path GEMM + attention -> tmp
    path_attn<<<N / 16, 256, ldsB, stream>>>(pp, Wp, index, hbuf, bpath, bA, tmp, N);
    // K_C: sparse adjacency scatter
    edge_scatter<<<2048, 256, 0, stream>>>(ei, WAT, tmp, E);
    // K_D: final projection
    final_out<<<(N + 255) / 256, 256, 0, stream>>>(tmp, Wend, bend, out, N);
}
